// MultiHeadAttention_75771813036481
// MI455X (gfx1250) — compile-verified, matrix operand order checked by reading, not run on hardware
//
#include <hip/hip_runtime.h>

namespace {

constexpr int kB  = 2;
constexpr int kS  = 2048;
constexpr int kD  = 768;
constexpr int kH  = 12;
constexpr int kDK = 64;
constexpr int kM  = kB * kS;            // 4096 token rows
constexpr float kScale = 0.125f;        // 1/sqrt(DK)

typedef __attribute__((ext_vector_type(16))) __bf16 v16bf;
typedef __attribute__((ext_vector_type(8)))  float  v8f;
typedef unsigned short u16;
typedef unsigned int   u32;

__device__ __forceinline__ u16 f2bf(float f) {
  u32 x = __builtin_bit_cast(u32, f);
  if ((x & 0x7f800000u) != 0x7f800000u)          // finite: round-to-nearest-even
    x += 0x7fffu + ((x >> 16) & 1u);
  return (u16)(x >> 16);
}

__device__ __forceinline__ v8f wmma_bf16(v16bf a, v16bf b, v8f c) {
  // v_wmma_f32_16x16x32_bf16: (neg_a, A, neg_b, B, c_mod, C, reuse_a, reuse_b)
  return __builtin_amdgcn_wmma_f32_16x16x32_bf16(false, a, false, b, (short)0, c,
                                                 false, false);
}

// ---- fragment loaders ------------------------------------------------------
// ISA 7.12.2, 16-bit A 16x32: lane L (row/col = L&15, half hl = L>>4) holds
// element i at K = (i<8 ? i : i+8) + 8*hl.  B (32x16) is the mirrored layout:
// lane holds column n = L&15 of B, i.e. row n of the [N,K] source matrix.

// Global version: buffers written by *earlier kernels* only -> safe dword loads.
__device__ __forceinline__ v16bf load_frag_g(const u16* __restrict__ base, int row,
                                             int stride, int kbase, int hl) {
  union { v16bf v; u32 d[8]; } f;
  const u16* p = base + (size_t)row * stride + kbase + (hl << 3);
#pragma unroll
  for (int g = 0; g < 8; ++g) {
    const int k = (g < 4) ? (g << 1) : (16 + ((g - 4) << 1));  // pairwise dwords
    f.d[g] = *(const u32*)(p + k);
  }
  return f.v;
}

// LDS version: same-kernel u16 stores precede these reads -> u16 loads (alias safe).
__device__ __forceinline__ v16bf load_frag_s(const u16* p0, int row, int stride,
                                             int hl) {
  union { v16bf v; u16 u[16]; } f;
  const u16* p = p0 + row * stride + (hl << 3);
#pragma unroll
  for (int i = 0; i < 16; ++i) f.u[i] = p[(i < 8) ? i : (i + 8)];
  return f.v;
}

// ---- elementwise helpers ---------------------------------------------------

__global__ void __launch_bounds__(256) k_f32_to_bf16(const float* __restrict__ in,
                                                     u16* __restrict__ out, int n) {
  const int i = blockIdx.x * 256 + threadIdx.x;
  if (i < n) out[i] = f2bf(in[i]);
}

// V [B*H][S][DK] (each head is a contiguous slab of the projected buffer)
//   -> Vt [B*H][DK][S] for contiguous score@V B-fragments.
__global__ void __launch_bounds__(256) k_head_transpose(const u16* __restrict__ v,
                                                        u16* __restrict__ vt) {
  const int i  = blockIdx.x * 256 + threadIdx.x;   // over B*H*DK*S (exact grid)
  const int s  = i & (kS - 1);
  const int dk = (i >> 11) & (kDK - 1);
  const int bh = i >> 17;
  vt[i] = v[((size_t)bh << 17) + (s << 6) + dk];
}

// ---- GEMM: out = A(bf16,[M=4096,K=768]) @ W^T(bf16,[768,768]) + bias -------
// One wave computes a 16x64 strip; 8 waves per block; 3072 wave-tasks total.
__global__ void __launch_bounds__(256) k_gemm_xwT(
    const u16* __restrict__ A, const u16* __restrict__ W,
    const float* __restrict__ bias, u16* __restrict__ out_bf,
    float* __restrict__ out_f32) {
  const int lane = threadIdx.x & 31;
  const int ln = lane & 15, hl = lane >> 4;
  const int wid = (blockIdx.x << 3) | (threadIdx.x >> 5);
  const int m0 = (wid / 12) << 4;            // 256 row strips
  const int n0 = (wid % 12) << 6;            // 12 col strips of 64

  v8f acc[4] = {};
  for (int kt = 0; kt < kD; kt += 32) {
    const v16bf a = load_frag_g(A, m0 + ln, kD, kt, hl);
#pragma unroll
    for (int g = 0; g < 4; ++g) {
      const v16bf b = load_frag_g(W, n0 + (g << 4) + ln, kD, kt, hl);
      acc[g] = wmma_bf16(a, b, acc[g]);
    }
  }
#pragma unroll
  for (int g = 0; g < 4; ++g) {
    const int n = n0 + (g << 4) + ln;
    const float bv = bias[n];
#pragma unroll
    for (int j = 0; j < 8; ++j) {
      const int m = m0 + j + (hl << 3);
      const float val = acc[g][j] + bv;
      if (out_f32) out_f32[(size_t)m * kD + n] = val;
      else         out_bf [(size_t)m * kD + n] = f2bf(val);
    }
  }
}

// ---- Pass 1: per (b,h) 16-row strip: online softmax stats + out = score@V --
__global__ void __launch_bounds__(256) k_attn_stats(
    const u16* __restrict__ Qp, const u16* __restrict__ Kp,
    const u16* __restrict__ Vt, u16* __restrict__ outb,
    float* __restrict__ rowmax, float* __restrict__ rowsum) {
  __shared__ u16 s_sc[8][16][34];            // per-wave 16x32 score tile (+pad)
  const int wblk = threadIdx.x >> 5;
  const int lane = threadIdx.x & 31;
  const int ln = lane & 15, hl = lane >> 4;
  const int task = (blockIdx.x << 3) | wblk; // 24 bh * 128 strips = 3072
  const int strip = task & 127;
  const int bh = task >> 7;
  const int b = bh / kH, h = bh % kH;
  const int m0 = strip << 4;

  const u16* Qh = Qp + (size_t)bh * kS * kDK;   // [S,64] row-major
  const u16* Kh = Kp + (size_t)bh * kS * kDK;
  const u16* Vh = Vt + (size_t)bh * kDK * kS;   // [64,S] row-major

  const v16bf qa0 = load_frag_g(Qh, m0 + ln, kDK, 0,  hl);
  const v16bf qa1 = load_frag_g(Qh, m0 + ln, kDK, 32, hl);

  v8f og[4] = {};
  float rmax[8], rsum[8];
#pragma unroll
  for (int j = 0; j < 8; ++j) { rmax[j] = -1e30f; rsum[j] = 0.f; }

  for (int pt = 0; pt < kS / 32; ++pt) {
    const int jn0 = pt << 5;
    if (pt + 1 < kS / 32)
      __builtin_prefetch(Kh + (size_t)(jn0 + 32 + ln) * kDK, 0, 3);

    // score tiles [16 x 32] = Q(16x64) @ K^T(64x32), scaled
    v8f s0 = {}, s1 = {};
    s0 = wmma_bf16(qa0, load_frag_g(Kh, jn0 + ln,      kDK, 0,  hl), s0);
    s0 = wmma_bf16(qa1, load_frag_g(Kh, jn0 + ln,      kDK, 32, hl), s0);
    s1 = wmma_bf16(qa0, load_frag_g(Kh, jn0 + 16 + ln, kDK, 0,  hl), s1);
    s1 = wmma_bf16(qa1, load_frag_g(Kh, jn0 + 16 + ln, kDK, 32, hl), s1);
#pragma unroll
    for (int j = 0; j < 8; ++j) { s0[j] *= kScale; s1[j] *= kScale; }

    // stash scaled scores (bf16) in LDS, D-layout -> row-major [16][32]
#pragma unroll
    for (int j = 0; j < 8; ++j) {
      s_sc[wblk][j + (hl << 3)][ln]      = f2bf(s0[j]);
      s_sc[wblk][j + (hl << 3)][16 + ln] = f2bf(s1[j]);
    }

    // online row max / sum-of-exp (reduce across the 16-lane half only)
#pragma unroll
    for (int j = 0; j < 8; ++j) {
      float tm = fmaxf(s0[j], s1[j]);
#pragma unroll
      for (int msk = 1; msk < 16; msk <<= 1) tm = fmaxf(tm, __shfl_xor(tm, msk));
      const float nm = fmaxf(rmax[j], tm);
      float e = __expf(s0[j] - nm) + __expf(s1[j] - nm);
#pragma unroll
      for (int msk = 1; msk < 16; msk <<= 1) e += __shfl_xor(e, msk);
      rsum[j] = rsum[j] * __expf(rmax[j] - nm) + e;
      rmax[j] = nm;
    }

    // out(16x64) += score(16x32) @ V(32x64)   (raw scores: the reference's bug)
    const v16bf sa = load_frag_s(&s_sc[wblk][0][0], ln, 34, hl);
#pragma unroll
    for (int g = 0; g < 4; ++g) {
      const v16bf vb = load_frag_g(Vh, (g << 4) + ln, kS, jn0, hl);
      og[g] = wmma_bf16(sa, vb, og[g]);
    }
  }

  // store head output directly in concat layout: [b][s2][h*64+dk], bf16
#pragma unroll
  for (int g = 0; g < 4; ++g) {
#pragma unroll
    for (int j = 0; j < 8; ++j) {
      const int s2 = m0 + j + (hl << 3);
      const int col = (g << 4) + ln;
      outb[((size_t)b * kS + s2) * kD + h * kDK + col] = f2bf(og[g][j]);
    }
  }
  if (ln == 0) {                       // lanes 0 and 16 own rows 0-7 / 8-15
#pragma unroll
    for (int j = 0; j < 8; ++j) {
      const size_t idx = (size_t)bh * kS + m0 + j + (hl << 3);
      rowmax[idx] = rmax[j];
      rowsum[idx] = rsum[j];
    }
  }
}

// ---- Pass 2: recompute scores (compute is free), write attn exactly once ---
__global__ void __launch_bounds__(256) k_attn_write(
    const u16* __restrict__ Qp, const u16* __restrict__ Kp,
    const float* __restrict__ rowmax, const float* __restrict__ rowsum,
    float* __restrict__ attn) {
  const int lane = threadIdx.x & 31;
  const int ln = lane & 15, hl = lane >> 4;
  const int task = (blockIdx.x << 3) | (threadIdx.x >> 5);
  const int strip = task & 127;
  const int bh = task >> 7;
  const int m0 = strip << 4;

  const u16* Qh = Qp + (size_t)bh * kS * kDK;
  const u16* Kh = Kp + (size_t)bh * kS * kDK;
  float* arow = attn + (size_t)bh * kS * kS;

  const v16bf qa0 = load_frag_g(Qh, m0 + ln, kDK, 0,  hl);
  const v16bf qa1 = load_frag_g(Qh, m0 + ln, kDK, 32, hl);

  float rm[8], ri[8];
#pragma unroll
  for (int j = 0; j < 8; ++j) {
    const size_t idx = (size_t)bh * kS + m0 + j + (hl << 3);
    rm[j] = rowmax[idx];
    ri[j] = 1.0f / rowsum[idx];
  }

  for (int jt = 0; jt < kS / 16; ++jt) {
    const int c0 = jt << 4;
    if (jt + 1 < kS / 16)
      __builtin_prefetch(Kh + (size_t)(c0 + 16 + ln) * kDK, 0, 3);
    v8f s = {};
    s = wmma_bf16(qa0, load_frag_g(Kh, c0 + ln, kDK, 0,  hl), s);
    s = wmma_bf16(qa1, load_frag_g(Kh, c0 + ln, kDK, 32, hl), s);
#pragma unroll
    for (int j = 0; j < 8; ++j) {
      const float val = __expf(s[j] * kScale - rm[j]) * ri[j];
      arow[(size_t)(m0 + j + (hl << 3)) * kS + c0 + ln] = val;  // coalesced
    }
  }
}

} // namespace

extern "C" void kernel_launch(void* const* d_in, const int* in_sizes, int n_in,
                              void* d_out, int out_size, void* d_ws, size_t ws_size,
                              hipStream_t stream) {
  const float* q  = (const float*)d_in[0];
  const float* k  = (const float*)d_in[1];
  const float* v  = (const float*)d_in[2];
  const float* Wq = (const float*)d_in[3];
  const float* bq = (const float*)d_in[4];
  const float* Wk = (const float*)d_in[5];
  const float* bk = (const float*)d_in[6];
  const float* Wv = (const float*)d_in[7];
  const float* bv = (const float*)d_in[8];
  const float* Wc = (const float*)d_in[9];
  const float* bc = (const float*)d_in[10];

  float* out  = (float*)d_out;                        // [B,S,D]
  float* attn = out + (size_t)kB * kS * kD;           // [B,H,S,S]

  // workspace carve (≈56 MB total)
  char* w = (char*)d_ws;
  auto take = [&](size_t bytes) {
    char* p = w;
    w += (bytes + 255) & ~(size_t)255;
    return p;
  };
  const size_t actE = (size_t)kM * kD;    // 3,145,728
  const size_t wE   = (size_t)kD * kD;    //   589,824
  u16*   qb   = (u16*)take(actE * 2);
  u16*   kb   = (u16*)take(actE * 2);
  u16*   vb   = (u16*)take(actE * 2);
  u16*   Wqb  = (u16*)take(wE * 2);
  u16*   Wkb  = (u16*)take(wE * 2);
  u16*   Wvb  = (u16*)take(wE * 2);
  u16*   Wcb  = (u16*)take(wE * 2);
  u16*   Qp   = (u16*)take(actE * 2);
  u16*   Kp   = (u16*)take(actE * 2);
  u16*   Vp   = (u16*)take(actE * 2);
  u16*   Vt   = (u16*)take(actE * 2);
  u16*   outb = (u16*)take(actE * 2);
  float* rmx  = (float*)take((size_t)kB * kH * kS * 4);
  float* rsm  = (float*)take((size_t)kB * kH * kS * 4);

  auto conv = [&](const float* src, u16* dst, size_t n) {
    k_f32_to_bf16<<<(unsigned)((n + 255) / 256), 256, 0, stream>>>(src, dst, (int)n);
  };
  conv(q, qb, actE);  conv(k, kb, actE);  conv(v, vb, actE);
  conv(Wq, Wqb, wE);  conv(Wk, Wkb, wE);  conv(Wv, Wvb, wE);  conv(Wc, Wcb, wE);

  // projections (bf16 out for downstream WMMA)
  k_gemm_xwT<<<384, 256, 0, stream>>>(qb, Wqb, bq, Qp, nullptr);
  k_gemm_xwT<<<384, 256, 0, stream>>>(kb, Wkb, bk, Kp, nullptr);
  k_gemm_xwT<<<384, 256, 0, stream>>>(vb, Wvb, bv, Vp, nullptr);

  // per-head V transpose for contiguous score@V B-fragments
  k_head_transpose<<<(unsigned)(actE / 256), 256, 0, stream>>>(Vp, Vt);

  // pass 1: softmax stats + head outputs (fused score@V, raw scores)
  k_attn_stats<<<384, 256, 0, stream>>>(Qp, Kp, Vt, outb, rmx, rsm);

  // pass 2: recompute scores, emit normalized attn (single 402 MB write)
  k_attn_write<<<384, 256, 0, stream>>>(Qp, Kp, rmx, rsm, attn);

  // final projection, f32 output
  k_gemm_xwT<<<384, 256, 0, stream>>>(outb, Wcb, bc, nullptr, out);
}
